// ParallelAttention_router_3332894622521
// MI455X (gfx1250) — compile-verified, matrix operand order checked by reading, not run on hardware
//
#include <hip/hip_runtime.h>
#include <hip/hip_bf16.h>

typedef __attribute__((ext_vector_type(16))) __bf16 v16bf;
typedef __attribute__((ext_vector_type(8)))  __bf16 v8bf;
typedef __attribute__((ext_vector_type(8)))  float  v8f;

#define HIDDEN 4096
#define NE     64      // experts
#define NC     192     // 3*NE output channels of fused QKV
#define KC     32      // K chunk per WMMA
#define MT     64      // tokens per workgroup
#define LDA    40      // lA stride in bf16 elems (32 + 8 pad -> 80B, 16B aligned)
#define LDB    40
#define LDM    200     // mix stride in floats (192 + 8 pad -> 800B, 16B aligned)

__device__ __forceinline__ unsigned short bf16_rte(float f) {
    unsigned u = __builtin_bit_cast(unsigned, f);
    unsigned r = u + 0x7FFFu + ((u >> 16) & 1u);   // round-to-nearest-even
    return (unsigned short)(r >> 16);
}

__global__ __launch_bounds__(256)
void moe_router_fused_kernel(const float* __restrict__ hidden,
                             const float* __restrict__ Wqkv,
                             float* __restrict__ out) {
    // 51200 bytes; lA/lB live during the K loop, lMix reuses the space after.
    __shared__ __align__(16) char smem[MT * LDM * 4];
    __bf16* lA   = (__bf16*)smem;                    // MT*LDA*2  = 5120 B
    __bf16* lB   = (__bf16*)(smem + MT * LDA * 2);   // NC*LDB*2  = 15360 B
    float*  lMix = (float*)smem;                     // MT*LDM*4  = 51200 B

    const int tid  = threadIdx.x;
    const int lane = tid & 31;
    const int wv   = tid >> 5;          // 8 waves
    const int t0   = blockIdx.x * MT;

    // 4 m-tiles x 12 n-tiles of 16x16; wave w -> m = w&3, n-tiles (w>>2)*6 + j
    const int mt      = wv & 3;
    const int nbase_w = (wv >> 2) * 6;

    v8f acc[6] = {};

    const int frow = (lane & 15);          // row within 16x16 tile this lane owns
    const int koff = (lane >> 4) * 8;      // lanes 0-15: K 0-7/16-23, lanes 16-31: K 8-15/24-31

    for (int kc = 0; kc < HIDDEN; kc += KC) {
        // ---- stage globals into registers (hidden streams from HBM, W from L2)
        float4 ra[2];
        #pragma unroll
        for (int i = 0; i < 2; ++i) {
            int idx = tid * 2 + i;                 // 0..511 = 64 rows x 8 float4
            int row = idx >> 3, q = idx & 7;
            ra[i] = *(const float4*)&hidden[(size_t)(t0 + row) * HIDDEN + kc + q * 4];
        }
        float4 rb[6];
        #pragma unroll
        for (int i = 0; i < 6; ++i) {
            int idx = tid + i * 256;               // 0..1535 = 192 rows x 8 float4
            int row = idx >> 3, q = idx & 7;
            rb[i] = *(const float4*)&Wqkv[(size_t)row * HIDDEN + kc + q * 4];
        }
        // prefetch the streaming A operand two chunks ahead (global_prefetch_b8)
        if (kc + 2 * KC < HIDDEN) {
            int idx = tid * 2;
            int row = idx >> 3, q = idx & 7;
            __builtin_prefetch(&hidden[(size_t)(t0 + row) * HIDDEN + kc + 2 * KC + q * 4], 0, 1);
        }

        __syncthreads();   // previous iteration's WMMA reads of LDS are done

        // ---- registers -> LDS as bf16
        #pragma unroll
        for (int i = 0; i < 2; ++i) {
            int idx = tid * 2 + i;
            int row = idx >> 3, q = idx & 7;
            ushort4 h;
            h.x = bf16_rte(ra[i].x); h.y = bf16_rte(ra[i].y);
            h.z = bf16_rte(ra[i].z); h.w = bf16_rte(ra[i].w);
            *(ushort4*)&lA[row * LDA + q * 4] = h;
        }
        #pragma unroll
        for (int i = 0; i < 6; ++i) {
            int idx = tid + i * 256;
            int row = idx >> 3, q = idx & 7;
            ushort4 h;
            h.x = bf16_rte(rb[i].x); h.y = bf16_rte(rb[i].y);
            h.z = bf16_rte(rb[i].z); h.w = bf16_rte(rb[i].w);
            *(ushort4*)&lB[row * LDB + q * 4] = h;
        }
        __syncthreads();

        // ---- WMMA compute: A + 6 B fragments, 6 MMAs.
        v16bf afrag;
        {
            const __bf16* ap = &lA[(mt * 16 + frow) * LDA + koff];
            v8bf a0 = *(const v8bf*)ap;            // K koff..koff+7
            v8bf a1 = *(const v8bf*)(ap + 16);     // K koff+16..koff+23
            #pragma unroll
            for (int i = 0; i < 8; ++i) { afrag[i] = a0[i]; afrag[i + 8] = a1[i]; }
        }
        v16bf bfrag[6];
        #pragma unroll
        for (int j = 0; j < 6; ++j) {
            const __bf16* bp = &lB[((nbase_w + j) * 16 + frow) * LDB + koff];
            v8bf b0 = *(const v8bf*)bp;
            v8bf b1 = *(const v8bf*)(bp + 16);
            #pragma unroll
            for (int i = 0; i < 8; ++i) { bfrag[j][i] = b0[i]; bfrag[j][i + 8] = b1[i]; }
        }
        #pragma unroll
        for (int j = 0; j < 6; ++j) {
            acc[j] = __builtin_amdgcn_wmma_f32_16x16x32_bf16(
                         false, afrag, false, bfrag[j], (short)0, acc[j], false, false);
        }
        // Scheduling directive: issue all 14 LDS reads (2 A + 12 B) as one
        // group, then the 6 WMMAs as one group.  This keeps all six B
        // fragments live in distinct registers so the DS reads pipeline and
        // the MMA chain drains behind a single s_wait_dscnt instead of one
        // LDS round-trip per MMA.
        __builtin_amdgcn_sched_group_barrier(0x100, 14, 0);  // 14x DS_READ
        __builtin_amdgcn_sched_group_barrier(0x008,  6, 0);  // 6x MFMA/WMMA
    }

    __syncthreads();   // lA/lB dead; reuse LDS for the mix tile

    // ---- spill accumulators to LDS mix[64][192]
    // C/D layout: VGPR i, lane l -> M = i + (l>=16 ? 8 : 0), N = l & 15
    {
        int Mb = mt * 16 + ((lane >> 4) ? 8 : 0);
        int Nc = lane & 15;
        #pragma unroll
        for (int j = 0; j < 6; ++j) {
            int n = (nbase_w + j) * 16 + Nc;
            #pragma unroll
            for (int i = 0; i < 8; ++i)
                lMix[(Mb + i) * LDM + n] = acc[j][i];
        }
    }
    __syncthreads();

    // ---- fused per-token softmax attention:
    // q=mix[:, 0:64), k=mix[:, 64:128), v=mix[:, 128:192)
    // out[t][e] = sum_f softmax_f(q_e * k_f) * v_f
    // wave wv handles tokens 8*wv .. 8*wv+7; each lane owns e = lane and e = lane+32
    #pragma unroll 1
    for (int tt = 0; tt < 8; ++tt) {
        int t = wv * 8 + tt;
        const float* row = &lMix[t * LDM];
        float q0 = row[lane];
        float q1 = row[lane + 32];

        float m0 = -3.0e38f, m1 = -3.0e38f;
        #pragma unroll
        for (int f = 0; f < NE; f += 4) {
            float4 kk = *(const float4*)&row[NE + f];       // broadcast LDS read
            m0 = fmaxf(m0, fmaxf(fmaxf(q0 * kk.x, q0 * kk.y), fmaxf(q0 * kk.z, q0 * kk.w)));
            m1 = fmaxf(m1, fmaxf(fmaxf(q1 * kk.x, q1 * kk.y), fmaxf(q1 * kk.z, q1 * kk.w)));
        }
        float s0 = 0.f, o0 = 0.f, s1 = 0.f, o1 = 0.f;
        #pragma unroll
        for (int f = 0; f < NE; f += 4) {
            float4 kk = *(const float4*)&row[NE + f];
            float4 vv = *(const float4*)&row[2 * NE + f];
            float e;
            e = __expf(q0 * kk.x - m0); s0 += e; o0 += e * vv.x;
            e = __expf(q0 * kk.y - m0); s0 += e; o0 += e * vv.y;
            e = __expf(q0 * kk.z - m0); s0 += e; o0 += e * vv.z;
            e = __expf(q0 * kk.w - m0); s0 += e; o0 += e * vv.w;
            e = __expf(q1 * kk.x - m1); s1 += e; o1 += e * vv.x;
            e = __expf(q1 * kk.y - m1); s1 += e; o1 += e * vv.y;
            e = __expf(q1 * kk.z - m1); s1 += e; o1 += e * vv.z;
            e = __expf(q1 * kk.w - m1); s1 += e; o1 += e * vv.w;
        }
        out[(size_t)(t0 + t) * NE + lane]      = o0 / s0;
        out[(size_t)(t0 + t) * NE + lane + 32] = o1 / s1;
    }
}

extern "C" void kernel_launch(void* const* d_in, const int* in_sizes, int n_in,
                              void* d_out, int out_size, void* d_ws, size_t ws_size,
                              hipStream_t stream) {
    (void)n_in; (void)out_size; (void)d_ws; (void)ws_size;
    const float* hidden = (const float*)d_in[0];
    const float* Wqkv   = (const float*)d_in[1];
    float* out          = (float*)d_out;

    int ntokens = in_sizes[0] / HIDDEN;          // 16384
    dim3 grid(ntokens / MT);                     // 256 workgroups
    dim3 block(256);                             // 8 wave32s
    hipLaunchKernelGGL(moe_router_fused_kernel, grid, block, 0, stream,
                       hidden, Wqkv, out);
}
